// DistFlashAttn_20925080666743
// MI455X (gfx1250) — matrix operand order repaired and checked by reading, not compiled, not run on hardware
//
#include <hip/hip_runtime.h>

// ---------------------------------------------------------------------------
// Ragged-range flash attention (local causal blocks + remote block) fused via
// online softmax, for MI455X (gfx1250, wave32, WMMA).
//   out : [Sq, H, D] fp32     lse : [H, Sq] fp32 (concatenated in d_out)
// Compute: bf16 v_wmma_f32_16x16x32_bf16 for QK^T, PV, and row-sum(P*ones);
// fp32 softmax state. Row-max reduced with batched ds_swizzle XOR butterflies.
// ---------------------------------------------------------------------------

typedef __attribute__((ext_vector_type(16))) __bf16 v16bf;
typedef __attribute__((ext_vector_type(8)))  __bf16 v8bf;
typedef __attribute__((ext_vector_type(4)))  __bf16 v4bf;
typedef __attribute__((ext_vector_type(8)))  float  v8f;

constexpr int kH   = 8;    // heads
constexpr int kD   = 64;   // head dim
constexpr int BQ   = 64;   // query tile per block
constexpr int BK   = 64;   // key tile
constexpr int NW   = 4;    // waves per block (wave32)
constexpr int MAXR = 8;

__device__ __forceinline__ __bf16 f2bf(float f) {
  unsigned u = __builtin_bit_cast(unsigned, f);
  u += 0x7FFFu + ((u >> 16) & 1u);            // round-to-nearest-even
  unsigned short hs = (unsigned short)(u >> 16);
  return __builtin_bit_cast(__bf16, hs);
}

// Build a v16bf WMMA operand from two contiguous 8-half (16B) LDS loads.
__device__ __forceinline__ v16bf lds16(const __bf16* p0, const __bf16* p1) {
  v8bf lo = *(const v8bf*)p0;
  v8bf hi = *(const v8bf*)p1;
  v16bf r;
#pragma unroll
  for (int i = 0; i < 8; ++i) { r[i] = lo[i]; r[i + 8] = hi[i]; }
  return r;
}

// One XOR-butterfly max step over 8 independent rows: 8 ds_swizzle issued
// back-to-back (single dscnt wait), then 8 v_max.
#define MAX_STEP(PATTERN)                                                     \
  {                                                                           \
    float t_[8];                                                              \
    _Pragma("unroll") for (int r_ = 0; r_ < 8; ++r_) t_[r_] =                 \
        __builtin_bit_cast(float, __builtin_amdgcn_ds_swizzle(                \
                                      __builtin_bit_cast(int, mx[r_]),        \
                                      (PATTERN)));                            \
    _Pragma("unroll") for (int r_ = 0; r_ < 8; ++r_) mx[r_] =                 \
        fmaxf(mx[r_], t_[r_]);                                                \
  }

__global__ __launch_bounds__(NW * 32, 1)
void ffa_fused_kernel(const float* __restrict__ q,
                      const float* __restrict__ kl, const float* __restrict__ vl,
                      const float* __restrict__ kr, const float* __restrict__ vr,
                      const int* __restrict__ qrl, const int* __restrict__ krl,
                      const int* __restrict__ czl, int nrl,
                      const int* __restrict__ qrr, const int* __restrict__ krr,
                      const int* __restrict__ czr, int nrr,
                      float* __restrict__ out, float* __restrict__ lse,
                      int Sq, int Sk) {
  __shared__ __align__(16) __bf16 sK [BK][kD];       // K tile, [key][d]
  __shared__ __align__(16) __bf16 sVt[kD][BK];       // V tile transposed, [d][key]
  __shared__ __align__(16) __bf16 sP [NW][16][BK];   // per-wave P staging

  const int tid  = threadIdx.x;
  const int w    = tid >> 5;        // wave id: query strip 16*w .. 16*w+15
  const int lane = tid & 31;
  const int lh   = lane >> 4;       // half-wave (0/1)
  const int ln   = lane & 15;
  const int q0   = blockIdx.x * BQ;
  const int h    = blockIdx.y;
  const int qbase = q0 + w * 16 + lh * 8;   // accumulator row r8 <-> query qbase+r8

  // ---- Q -> A fragments (two K=32 chunks over D=64), 1/sqrt(D) folded in --
  const float scl  = 0.125f;
  const int   qrow = q0 + w * 16 + ln;
  const float* qp  = q + ((size_t)qrow * kH + h) * kD;
  v16bf qfrag[2];
#pragma unroll
  for (int c = 0; c < 2; ++c) {
#pragma unroll
    for (int i = 0; i < 8; ++i) {
      // A layout: VGPR i holds K = (i<4 ? 2i : 16+2(i-4)) + 8*half (+32c), pair
      int d = ((i < 4) ? 2 * i : 16 + 2 * (i - 4)) + lh * 8 + c * 32;
      float2 qv = *(const float2*)(qp + d);
      qfrag[c][2 * i]     = f2bf(qv.x * scl);
      qfrag[c][2 * i + 1] = f2bf(qv.y * scl);
    }
  }

  // all-ones B operand for row-sum-by-WMMA
  v16bf ones;
#pragma unroll
  for (int i = 0; i < 16; ++i) ones[i] = __builtin_bit_cast(__bf16, (unsigned short)0x3F80u);

  // ---- running flash-attention state -------------------------------------
  v8f oacc[4];
#pragma unroll
  for (int t = 0; t < 4; ++t)
#pragma unroll
    for (int r = 0; r < 8; ++r) oacc[t][r] = 0.f;
  float run_max[8], run_sum[8];
#pragma unroll
  for (int r = 0; r < 8; ++r) { run_max[r] = -1e30f; run_sum[r] = 0.f; }

  // ---- two partials (local, remote) through ONE code body -----------------
  const float* KP[2] = {kl, kr};
  const float* VP[2] = {vl, vr};
  const int*   QR[2] = {qrl, qrr};
  const int*   KR[2] = {krl, krr};
  const int*   CZ[2] = {czl, czr};
  const int    NR[2] = {nrl, nrr};

  for (int pass = 0; pass < 2; ++pass) {
    const float* kp = KP[pass];
    const float* vp = VP[pass];
    const int    nr = NR[pass];
    int qs[MAXR], qe[MAXR], ks[MAXR], ke[MAXR], co[MAXR], cs[MAXR];
    int kmin = Sk, kmax = 0;
    for (int r = 0; r < nr; ++r) {
      qs[r] = QR[pass][2 * r];  qe[r] = QR[pass][2 * r + 1];
      ks[r] = KR[pass][2 * r];  ke[r] = KR[pass][2 * r + 1];
      cs[r] = (CZ[pass][r] != 0);
      // causal: (k-ks) <= (q-qs) + (ke-ks)-(qe-qs)  =>  k <= q + co
      co[r] = ks[r] - qs[r] + (ke[r] - ks[r]) - (qe[r] - qs[r]);
      kmin = min(kmin, ks[r]); kmax = max(kmax, ke[r]);
    }

    for (int k0 = (kmin / BK) * BK; k0 < kmax; k0 += BK) {
      // ---- uniform tile classification (SALU) ---------------------------
      bool any = false, full = false;
      for (int r = 0; r < nr; ++r) {
        int loq = max(q0, qs[r]), hiq = min(q0 + BQ, qe[r]);
        int lok = max(k0, ks[r]), hik = min(k0 + BK, ke[r]);
        if (loq < hiq && lok < hik) {
          if (!cs[r] || (lok <= (hiq - 1) + co[r])) any = true;
          if (qs[r] <= q0 && qe[r] >= q0 + BQ && ks[r] <= k0 && ke[r] >= k0 + BK &&
              (!cs[r] || (k0 + BK - 1 <= q0 + co[r])))
            full = true;
        }
      }
      if (!any) continue;                      // uniform branch

      __syncthreads();                         // protect previous tile's LDS
      // ---- stage K and V^T tiles into LDS as bf16 -----------------------
      {
        int key = tid >> 1, dh = (tid & 1) * 32;
        const float* krow = kp + ((size_t)(k0 + key) * kH + h) * kD + dh;
        const float* vrow = vp + ((size_t)(k0 + key) * kH + h) * kD + dh;
        if (k0 + BK < kmax) {                  // pull next tile toward cache
          __builtin_prefetch(krow + (size_t)BK * kH * kD, 0, 1);
          __builtin_prefetch(vrow + (size_t)BK * kH * kD, 0, 1);
        }
#pragma unroll
        for (int j = 0; j < 32; j += 4) {
          float4 k4 = *(const float4*)(krow + j);
          v4bf kb;
          kb[0] = f2bf(k4.x); kb[1] = f2bf(k4.y);
          kb[2] = f2bf(k4.z); kb[3] = f2bf(k4.w);
          *(v4bf*)&sK[key][dh + j] = kb;
          float4 v4 = *(const float4*)(vrow + j);
          sVt[dh + j + 0][key] = f2bf(v4.x);
          sVt[dh + j + 1][key] = f2bf(v4.y);
          sVt[dh + j + 2][key] = f2bf(v4.z);
          sVt[dh + j + 3][key] = f2bf(v4.w);
        }
      }
      __syncthreads();

      // ---- S = (Q*scale) K^T : 8 WMMAs per wave -------------------------
      v8f acc[4];
#pragma unroll
      for (int t = 0; t < 4; ++t) {
        v8f a;
#pragma unroll
        for (int r = 0; r < 8; ++r) a[r] = 0.f;
#pragma unroll
        for (int c = 0; c < 2; ++c) {
          // B layout: lane = key (N), K(d) = 32c + 8*half + {0..7, 16..23}
          const __bf16* bp = &sK[t * 16 + ln][c * 32 + lh * 8];
          v16bf b = lds16(bp, bp + 16);
          a = __builtin_amdgcn_wmma_f32_16x16x32_bf16(false, qfrag[c], false, b,
                                                      (short)0, a, false, false);
        }
        acc[t] = a;
      }

      // ---- ragged/causal masking: branchless 8-bit row-interval mask ----
      if (!full) {
#pragma unroll
        for (int t = 0; t < 4; ++t) {
          int kk = k0 + t * 16 + ln;           // this lane's key column
          unsigned rowmask = 0u;
          for (int r = 0; r < nr; ++r) {
            // allowed rows r8 form an interval [lo, hi) for fixed kk
            int lo = qs[r] - qbase;
            if (cs[r]) lo = max(lo, kk - co[r] - qbase);
            lo = min(max(lo, 0), 8);
            int hi = min(max(qe[r] - qbase, 0), 8);
            unsigned m = (lo < hi) ? ((1u << hi) - (1u << lo)) : 0u;
            bool k_ok = (kk >= ks[r]) & (kk < ke[r]);
            rowmask |= k_ok ? m : 0u;
          }
#pragma unroll
          for (int r8 = 0; r8 < 8; ++r8)
            acc[t][r8] = ((rowmask >> r8) & 1u) ? acc[t][r8] : -1e30f;
        }
      }

      // ---- online softmax: row-max via batched ds_swizzle butterflies ---
      float mx[8];
#pragma unroll
      for (int r = 0; r < 8; ++r)
        mx[r] = fmaxf(fmaxf(acc[0][r], acc[1][r]), fmaxf(acc[2][r], acc[3][r]));
      MAX_STEP(0x041F)   // SWAPX1
      MAX_STEP(0x081F)   // SWAPX2
      MAX_STEP(0x101F)   // SWAPX4
      MAX_STEP(0x201F)   // SWAPX8

      float nmax[8], scale[8];
#pragma unroll
      for (int r = 0; r < 8; ++r) {
        nmax[r]    = fmaxf(run_max[r], mx[r]);
        scale[r]   = __expf(run_max[r] - nmax[r]);
        run_max[r] = nmax[r];
      }
      // p = exp(s - nmax) (0 for masked), C-layout -> per-wave LDS staging
#pragma unroll
      for (int t = 0; t < 4; ++t) {
#pragma unroll
        for (int r = 0; r < 8; ++r) {
          float s = acc[t][r];
          float p = (s < -1e29f) ? 0.f : __expf(s - nmax[r]);
          sP[w][r + lh * 8][t * 16 + ln] = f2bf(p);
        }
      }
#pragma unroll
      for (int r = 0; r < 8; ++r)
#pragma unroll
        for (int t = 0; t < 4; ++t) oacc[t][r] *= scale[r];

      // ---- O += P V, and rowsum(P) = P x ones : 10 WMMAs per wave -------
      v8f rs;
#pragma unroll
      for (int r = 0; r < 8; ++r) rs[r] = 0.f;
#pragma unroll
      for (int c = 0; c < 2; ++c) {
        int kb = c * 32 + lh * 8;              // A-layout K base for this half
        v16bf ap = lds16(&sP[w][ln][kb], &sP[w][ln][kb + 16]);
        rs = __builtin_amdgcn_wmma_f32_16x16x32_bf16(false, ap, false, ones,
                                                     (short)0, rs, false, false);
#pragma unroll
        for (int t = 0; t < 4; ++t) {
          const __bf16* bp = &sVt[t * 16 + ln][c * 32 + lh * 8];
          v16bf bv = lds16(bp, bp + 16);
          oacc[t] = __builtin_amdgcn_wmma_f32_16x16x32_bf16(false, ap, false, bv,
                                                            (short)0, oacc[t],
                                                            false, false);
        }
      }
      // every column of rs holds the row-sum; fold into running denominator
#pragma unroll
      for (int r = 0; r < 8; ++r)
        run_sum[r] = run_sum[r] * scale[r] + rs[r];
    }
  }

  // ---- finalize: out = O / sum, lse = max + log(sum) ----------------------
#pragma unroll
  for (int r = 0; r < 8; ++r) {
    int   qq  = qbase + r;
    float inv = (run_sum[r] > 0.f) ? 1.f / run_sum[r] : 0.f;
    float* op = out + ((size_t)qq * kH + h) * kD;
#pragma unroll
    for (int t = 0; t < 4; ++t) op[t * 16 + ln] = oacc[t][r] * inv;
    if (ln == 0)
      lse[(size_t)h * Sq + qq] =
          (run_sum[r] > 0.f) ? (run_max[r] + __logf(run_sum[r])) : -1e30f;
  }
}

extern "C" void kernel_launch(void* const* d_in, const int* in_sizes, int n_in,
                              void* d_out, int out_size, void* d_ws, size_t ws_size,
                              hipStream_t stream) {
  const float* q  = (const float*)d_in[0];
  const float* kl = (const float*)d_in[1];
  const float* vl = (const float*)d_in[2];
  const float* kr = (const float*)d_in[3];
  const float* vr = (const float*)d_in[4];
  const int* qrl  = (const int*)d_in[5];
  const int* krl  = (const int*)d_in[6];
  const int* czl  = (const int*)d_in[7];
  const int* qrr  = (const int*)d_in[8];
  const int* krr  = (const int*)d_in[9];
  const int* czr  = (const int*)d_in[10];

  const int Sq  = in_sizes[0] / (kH * kD);
  const int Sk  = in_sizes[1] / (kH * kD);
  const int nrl = in_sizes[7];
  const int nrr = in_sizes[10];

  float* out = (float*)d_out;
  float* lse = out + (size_t)Sq * kH * kD;

  dim3 grid(Sq / BQ, kH);
  ffa_fused_kernel<<<grid, NW * 32, 0, stream>>>(q, kl, vl, kr, vr,
                                                 qrl, krl, czl, nrl,
                                                 qrr, krr, czr, nrr,
                                                 out, lse, Sq, Sk);
}
